// atten_75771813036289
// MI455X (gfx1250) — compile-verified, matrix operand order checked by reading, not run on hardware
//
#include <hip/hip_runtime.h>
#include <hip/hip_bf16.h>

// ---------------------------------------------------------------------------
// Phase 0: pack W (f32, row-major) into bf16 WMMA B-fragment layout (d_ws).
// Phase 1: atty = tanh(conrep @ W + b) via v_wmma_f32_16x16x32_bf16,
//          64 rows x 16 cols per wave (4 WMMA tiles share one B fragment).
// Phase 2: per-edge gather/dot/sigmoid/scale + atomic scatter-add into out.
// ---------------------------------------------------------------------------

typedef __attribute__((ext_vector_type(16))) __bf16 v16bf;
typedef __attribute__((ext_vector_type(8)))  float  v8f;
typedef __attribute__((ext_vector_type(4)))  float  v4f;

#define DIM 300
#define KSTEPS 10                     // ceil(300/32)
#define NTILES 19                     // ceil(300/16)

__device__ __forceinline__ __bf16 f2bf(float x) { return (__bf16)x; }

// --------------------------------------------------------------------------
// Pack W[k][n] into per-(ntile,kstep,lane) contiguous 16-element bf16
// fragments matching the 16-bit A/B 16x32 WMMA operand layout:
//   lane L: col = L%16, khalf = L/16
//   elems 0..7  -> K = ks*32 + khalf*8 + j
//   elems 8..15 -> K = ks*32 + 16 + khalf*8 + j
// Out-of-range K or col -> 0 (exact zero padding, done once).
// --------------------------------------------------------------------------
__global__ __launch_bounds__(32) void pack_W_kernel(
    const float* __restrict__ W, __bf16* __restrict__ Bpack)
{
    const int ks    = blockIdx.x % KSTEPS;
    const int ntile = blockIdx.x / KSTEPS;
    const int lane  = threadIdx.x;
    const int col   = ntile * 16 + (lane & 15);
    const int khalf = lane >> 4;
    const int k1    = ks * 32 + khalf * 8;
    const int k2    = k1 + 16;

    v16bf frag;
    #pragma unroll
    for (int j = 0; j < 8; ++j) {
        const int ka = k1 + j, kc = k2 + j;
        const float f1 = (col < DIM && ka < DIM) ? W[(size_t)ka * DIM + col] : 0.f;
        const float f2 = (col < DIM && kc < DIM) ? W[(size_t)kc * DIM + col] : 0.f;
        frag[j]     = f2bf(f1);
        frag[8 + j] = f2bf(f2);
    }
    v16bf* dst = (v16bf*)Bpack + ((size_t)ntile * KSTEPS + ks) * 32 + lane;
    *dst = frag;
}

// --------------------------------------------------------------------------
// GEMM: one wave per 64(M) x 16(N) output block = 4 WMMA tiles sharing B.
// A loads: unconditional float4 pairs (clamped row), tail K-step uses
// clamped-address loads + selects (no divergent branches anywhere hot).
// --------------------------------------------------------------------------
__global__ __launch_bounds__(32) void atty_gemm_kernel(
    const float* __restrict__ conrep,   // [n_nodes, DIM]
    const __bf16* __restrict__ Bpack,   // packed W fragments
    const float* __restrict__ bias,     // [DIM]
    float* __restrict__ atty,           // [n_nodes, DIM]
    int n_nodes)
{
    const int tile   = blockIdx.x;
    const int mblk   = tile / NTILES;          // 64-row block index
    const int ntile  = tile - mblk * NTILES;

    const int lane   = threadIdx.x;
    const int r      = lane & 15;
    const int khalf  = lane >> 4;
    const int col    = ntile * 16 + r;
    const bool col_ok = (col < DIM);

    // Clamped row pointers (clamped rows only affect their own masked stores).
    const float* Arow[4];
    #pragma unroll
    for (int s = 0; s < 4; ++s) {
        int row = mblk * 64 + s * 16 + r;
        row = row < n_nodes ? row : n_nodes - 1;
        Arow[s] = conrep + (size_t)row * DIM;
    }

    const v16bf* Bp = (const v16bf*)Bpack + (size_t)ntile * KSTEPS * 32 + lane;

    v8f acc[4];
    #pragma unroll
    for (int s = 0; s < 4; ++s)
        acc[s] = (v8f){0.f, 0.f, 0.f, 0.f, 0.f, 0.f, 0.f, 0.f};

    #pragma unroll
    for (int ks = 0; ks < KSTEPS; ++ks) {
        const int kb = ks * 32;
        const int k1 = kb + khalf * 8;          // elems 0..7
        const int k2 = k1 + 16;                 // elems 8..15

        const v16bf bfrag = Bp[ks * 32];        // one 32B contiguous load

        #pragma unroll
        for (int s = 0; s < 4; ++s) {
            v16bf afrag;
            if (kb + 31 < DIM) {
                // Fast path: max index = kb+8+16+7 = kb+31 < 300. Vector loads.
                const v4f lo0 = *(const v4f*)(Arow[s] + k1);
                const v4f lo1 = *(const v4f*)(Arow[s] + k1 + 4);
                const v4f hi0 = *(const v4f*)(Arow[s] + k2);
                const v4f hi1 = *(const v4f*)(Arow[s] + k2 + 4);
                #pragma unroll
                for (int j = 0; j < 4; ++j) {
                    afrag[j]      = f2bf(lo0[j]);
                    afrag[4 + j]  = f2bf(lo1[j]);
                    afrag[8 + j]  = f2bf(hi0[j]);
                    afrag[12 + j] = f2bf(hi1[j]);
                }
            } else {
                // Tail: clamped addresses + value selects (branch-free).
                #pragma unroll
                for (int j = 0; j < 8; ++j) {
                    const int ka = k1 + j, kc = k2 + j;
                    float a1 = Arow[s][ka < DIM ? ka : DIM - 1];
                    float a2 = Arow[s][kc < DIM ? kc : DIM - 1];
                    a1 = (ka < DIM) ? a1 : 0.f;
                    a2 = (kc < DIM) ? a2 : 0.f;
                    afrag[j]     = f2bf(a1);
                    afrag[8 + j] = f2bf(a2);
                }
            }
            acc[s] = __builtin_amdgcn_wmma_f32_16x16x32_bf16(
                false, afrag, false, bfrag, (short)0, acc[s], false, false);
        }
    }

    const float bv = bias[col < DIM ? col : DIM - 1];
    if (col_ok) {
        #pragma unroll
        for (int s = 0; s < 4; ++s) {
            #pragma unroll
            for (int v = 0; v < 8; ++v) {
                const int m = mblk * 64 + s * 16 + khalf * 8 + v;
                if (m < n_nodes)
                    atty[(size_t)m * DIM + col] = tanhf(acc[s][v] + bv);
            }
        }
    }
}

// --------------------------------------------------------------------------
// One wave32 per edge. messages streamed with nontemporal loads (read-once,
// 300 MB) so L2 keeps atty + out resident. Wave shuffle dot-reduction,
// global f32 atomics for the segment-sum.
// --------------------------------------------------------------------------
__global__ __launch_bounds__(256) void edge_gate_scatter_kernel(
    const float* __restrict__ messages, // [n_edges, DIM]
    const float* __restrict__ atty,     // [n_nodes, DIM]
    const int*   __restrict__ dst_idx,  // [n_edges]
    float* __restrict__ out,            // [n_nodes, DIM] (pre-zeroed)
    int n_edges)
{
    const int wave = blockIdx.x * (blockDim.x >> 5) + (threadIdx.x >> 5);
    if (wave >= n_edges) return;                 // uniform per wave
    const int lane = threadIdx.x & 31;

    const int dst = dst_idx[wave];
    const float* __restrict__ m = messages + (size_t)wave * DIM;
    const float* __restrict__ a = atty     + (size_t)dst  * DIM;

    float vals[10];
    float partial = 0.f;
    #pragma unroll
    for (int i = 0; i < 10; ++i) {
        const int k = lane + 32 * i;
        const float mv = (k < DIM) ? __builtin_nontemporal_load(m + k) : 0.f;
        const float av = (k < DIM) ? a[k] : 0.f;
        vals[i] = mv;
        partial += mv * av;
    }

    #pragma unroll
    for (int off = 16; off > 0; off >>= 1)
        partial += __shfl_xor(partial, off, 32);

    const float gate = 1.f / (1.f + __expf(-partial));

    float* __restrict__ o = out + (size_t)dst * DIM;
    #pragma unroll
    for (int i = 0; i < 10; ++i) {
        const int k = lane + 32 * i;
        if (k < DIM)
            atomicAdd(&o[k], gate * vals[i]);    // global_atomic_add_f32
    }
}

extern "C" void kernel_launch(void* const* d_in, const int* in_sizes, int n_in,
                              void* d_out, int out_size, void* d_ws, size_t ws_size,
                              hipStream_t stream) {
    const float* messages = (const float*)d_in[0];   // [E, 300]
    const float* conrep   = (const float*)d_in[1];   // [N, 300]
    const float* W        = (const float*)d_in[2];   // [300, 300]
    const float* bias     = (const float*)d_in[3];   // [300]
    const int*   dst_idx  = (const int*)  d_in[4];   // [E]
    float*       out      = (float*)d_out;           // [N, 300]

    const int n_edges = in_sizes[4];
    const int n_nodes = in_sizes[1] / DIM;

    // Workspace: [atty f32 N*300][Bpack bf16 fragments]
    float*  atty  = (float*)d_ws;
    __bf16* Bpack = (__bf16*)((char*)d_ws + (size_t)n_nodes * DIM * sizeof(float));

    // out accumulates via atomics -> must start at zero every call.
    hipMemsetAsync(out, 0, (size_t)out_size * sizeof(float), stream);

    // Phase 0: swizzle W into WMMA B-fragment layout (once, ~190 KB).
    pack_W_kernel<<<NTILES * KSTEPS, 32, 0, stream>>>(W, Bpack);

    // Phase 1: WMMA GEMM + bias + tanh. One wave per 64x16 output block.
    const int mblks = (n_nodes + 63) / 64;
    atty_gemm_kernel<<<mblks * NTILES, 32, 0, stream>>>(conrep, Bpack, bias, atty, n_nodes);

    // Phase 2: edge gather/gate/scatter (1 wave per edge, 8 waves per block).
    const int waves_per_block = 256 / 32;
    const int blocks = (n_edges + waves_per_block - 1) / waves_per_block;
    edge_gate_scatter_kernel<<<blocks, 256, 0, stream>>>(messages, atty, dst_idx, out, n_edges);
}
